// SelfAttentionBlock_9672266351200
// MI455X (gfx1250) — matrix operand order repaired and checked by reading, not compiled, not run on hardware
//
#include <hip/hip_runtime.h>

#define DEV __device__ __forceinline__

#define BATCH  2
#define SEQ    2048
#define DMODEL 1024
#define NHEADS 16
#define DHEAD  64
#define CONDF  768
#define ROWS   (BATCH * SEQ)   // 4096
#define NQKV   (3 * DMODEL)    // 3072

#if defined(__has_builtin)
#if __has_builtin(__builtin_amdgcn_tensor_load_to_lds) && \
    __has_builtin(__builtin_amdgcn_s_wait_tensorcnt)
#define HAVE_TDM 1
#endif
#endif

typedef __attribute__((ext_vector_type(16))) __bf16 v16bf;
typedef __attribute__((ext_vector_type(8)))  float  v8f;
typedef unsigned int u32x4 __attribute__((ext_vector_type(4)));
typedef int          i32x8 __attribute__((ext_vector_type(8)));
typedef int          i32x4 __attribute__((ext_vector_type(4)));

union FragB16 {
    v16bf v;
    uint4 u4[2];
    unsigned short us[16];
};

DEV unsigned short f2bf(float f) {
    union { float f; unsigned u; } c; c.f = f;
    unsigned u = c.u;
    u += 0x7FFFu + ((u >> 16) & 1u);   // round-to-nearest-even
    return (unsigned short)(u >> 16);
}
DEV float bf2f(unsigned short h) {
    union { unsigned u; float f; } c; c.u = ((unsigned)h) << 16;
    return c.f;
}

DEV v8f wmma_bf16(v16bf a, v16bf b, v8f c) {
    return __builtin_amdgcn_wmma_f32_16x16x32_bf16(
        false, a, false, b, (short)0, c, false, false);
}

// A-matrix fragment (16x32 bf16) from row-major [.., ld] at (row0, k0).
// Contiguous k order (same as B): lane<16: M=lane, K = k0+0..15 ; lane>=16: K = k0+16..31.
DEV v16bf load_a_frag(const unsigned short* base, int ld, int row0, int k0, int lane) {
    FragB16 f;
    int r  = row0 + (lane & 15);
    int kb = k0 + ((lane >> 4) << 4);
    const unsigned short* p = base + (size_t)r * ld + kb;
    f.u4[0] = *(const uint4*)(p);
    f.u4[1] = *(const uint4*)(p + 8);
    return f.v;
}

// B-matrix fragment (32x16 bf16), B[k][n] sourced from W[n][k] row-major.
// Contiguous k order (same as A): lane<16: N=lane, K = k0+0..15 ; lane>=16: K = k0+16..31.
DEV v16bf load_b_frag(const unsigned short* base, int ld, int n0, int k0, int lane) {
    FragB16 f;
    int n  = n0 + (lane & 15);
    int kb = k0 + ((lane >> 4) << 4);
    const unsigned short* p = base + (size_t)n * ld + kb;
    f.u4[0] = *(const uint4*)(p);
    f.u4[1] = *(const uint4*)(p + 8);
    return f.v;
}

#ifdef HAVE_TDM
// Issue a TDM 2-D tile load: global (row-major, 2-byte elems, row stride
// `stride0` elems) -> LDS at byte offset lds_off. tile_d0 = row length (elems),
// tile_d1 = #rows. Descriptor layout per CDNA5 ISA 8.3/8.4.
// clang-23 builtin arity: (u32x4, i32x8, i32x4, i32x4, i32x8, i32 cpol).
DEV void tdm_load_2d(const unsigned short* gsrc, unsigned lds_off,
                     unsigned tile_d0, unsigned tile_d1, unsigned stride0) {
    unsigned long long ga = (unsigned long long)gsrc;
    u32x4 g0 = {};
    g0[0] = 1u;                                    // count=1, normal mode
    g0[1] = lds_off;                               // lds_addr (bytes)
    g0[2] = (unsigned)ga;                          // global_addr[31:0]
    g0[3] = (unsigned)((ga >> 32) & 0x01FFFFFFu)   // global_addr[56:32]
          | (2u << 30);                            // type = 2 ("image")
    i32x8 g1 = {};
    g1[0] = (int)(1u << 16);                       // wg_mask=0, data_size=2B
    g1[1] = (int)((tile_d0 & 0xFFFFu) << 16);      // tensor_dim0 = tile_d0
    g1[2] = (int)((tile_d0 >> 16) | ((tile_d1 & 0xFFFFu) << 16)); // tensor_dim1
    g1[3] = (int)((tile_d1 >> 16) | (tile_d0 << 16));             // tile_dim0
    g1[4] = (int)tile_d1;                          // tile_dim1 (tile_dim2=0)
    g1[5] = (int)stride0;                          // tensor_dim0_stride[31:0]
    g1[6] = 0;
    g1[7] = 0;
    i32x4 g2 = {};
    i32x4 g3 = {};
    i32x8 g4 = {};
    __builtin_amdgcn_tensor_load_to_lds(g0, g1, g2, g3, g4, 0);
}
#endif

// ---------------------------------------------------------------- kernels ---

__global__ void k_cond_scale(const float* __restrict__ cond,
                             const float* __restrict__ w_norm,
                             float* __restrict__ cs) {
    int idx = blockIdx.x * blockDim.x + threadIdx.x;
    if (idx >= BATCH * DMODEL) return;
    int b = idx / DMODEL, d = idx % DMODEL;
    const float* c = cond + (size_t)b * CONDF;
    const float* w = w_norm + (size_t)d * CONDF;
    float acc = 0.f;
    for (int i = 0; i < CONDF; i += 4) {
        float4 cv = *(const float4*)(c + i);
        float4 wv = *(const float4*)(w + i);
        acc += cv.x * wv.x + cv.y * wv.y + cv.z * wv.z + cv.w * wv.w;
    }
    cs[idx] = acc + 1.f;
}

__global__ void k_f32_to_bf16(const float* __restrict__ in,
                              unsigned short* __restrict__ out, int n) {
    int i = blockIdx.x * blockDim.x + threadIdx.x;
    if (i < n) out[i] = f2bf(in[i]);
}

__global__ void k_rmsnorm(const float* __restrict__ x,
                          const float* __restrict__ cs,
                          unsigned short* __restrict__ xn) {
    int row = blockIdx.x;                  // 0..ROWS-1
    int b = row / SEQ;
    const float* xr = x + (size_t)row * DMODEL;
    float s = 0.f;
    for (int i = threadIdx.x; i < DMODEL; i += blockDim.x) {
        float v = xr[i]; s += v * v;
    }
    for (int o = 16; o > 0; o >>= 1) s += __shfl_xor(s, o, 32);
    __shared__ float red[8];
    if ((threadIdx.x & 31) == 0) red[threadIdx.x >> 5] = s;
    __syncthreads();
    float tot = 0.f;
    #pragma unroll
    for (int w = 0; w < 8; w++) tot += red[w];
    float inv = rsqrtf(tot * (1.0f / DMODEL) + 1e-6f);
    const float* csb = cs + (size_t)b * DMODEL;
    for (int i = threadIdx.x; i < DMODEL; i += blockDim.x)
        xn[(size_t)row * DMODEL + i] = f2bf(xr[i] * csb[i] * inv);
}

// Generic bf16 GEMM: C[M,N] (bf16) = A[M,K] * W[N,K]^T.
// One wave computes a 16x64 strip. All fragments are loaded into independent
// values before the 4 WMMAs so loads stay in flight across the math.
__global__ void k_gemm_bf16(const unsigned short* __restrict__ A,
                            const unsigned short* __restrict__ W,
                            unsigned short* __restrict__ C,
                            int M, int N, int K) {
    int wave = (blockIdx.x * blockDim.x + threadIdx.x) >> 5;
    int lane = threadIdx.x & 31;
    int n4 = N >> 6;
    int mt = wave / n4, ng = wave % n4;
    if (mt >= (M >> 4)) return;
    int n0 = ng * 64;
    v8f a0 = {}, a1 = {}, a2 = {}, a3 = {};
    #pragma unroll 4
    for (int k0 = 0; k0 < K; k0 += 32) {
        if (k0 + 64 < K) {   // prefetch ahead on the A stream
            __builtin_prefetch(A + (size_t)(mt * 16 + (lane & 15)) * K + k0 + 64, 0, 1);
        }
        v16bf a  = load_a_frag(A, K, mt * 16, k0, lane);
        v16bf b0 = load_b_frag(W, K, n0,      k0, lane);
        v16bf b1 = load_b_frag(W, K, n0 + 16, k0, lane);
        v16bf b2 = load_b_frag(W, K, n0 + 32, k0, lane);
        v16bf b3 = load_b_frag(W, K, n0 + 48, k0, lane);
        a0 = wmma_bf16(a, b0, a0);
        a1 = wmma_bf16(a, b1, a1);
        a2 = wmma_bf16(a, b2, a2);
        a3 = wmma_bf16(a, b3, a3);
    }
    int half = lane >> 4, col = n0 + (lane & 15);
    #pragma unroll
    for (int j = 0; j < 8; j++) {
        size_t row = (size_t)(mt * 16 + j + half * 8);
        C[row * N + col]      = f2bf(a0[j]);
        C[row * N + col + 16] = f2bf(a1[j]);
        C[row * N + col + 32] = f2bf(a2[j]);
        C[row * N + col + 48] = f2bf(a3[j]);
    }
}

// Per-head L2 normalize (x sqrt(scale)) + partial RoPE; scatter q,k [B,H,S,E]
// and v transposed [B,H,E,S]. One wave per (b,s,h); lane owns elems e, e+32.
__global__ void k_qkv_norm_rope(const unsigned short* __restrict__ qkv,
                                const float* __restrict__ pos,
                                const float* __restrict__ scale,
                                unsigned short* __restrict__ q,
                                unsigned short* __restrict__ k,
                                unsigned short* __restrict__ vT) {
    int wid  = (blockIdx.x * blockDim.x + threadIdx.x) >> 5;
    int lane = threadIdx.x & 31;
    int h   = wid % NHEADS;
    int row = wid / NHEADS;               // b*SEQ + s
    int b = row / SEQ, s = row % SEQ;
    float p  = pos[row];
    float ss = sqrtf(scale[h]);
    const unsigned short* base = qkv + (size_t)row * NQKV + h * DHEAD;

    #pragma unroll
    for (int t = 0; t < 2; t++) {          // q then k
        const unsigned short* src = base + t * DMODEL;
        float e0 = bf2f(src[lane]);
        float e1 = bf2f(src[lane + 32]);
        float sq = e0 * e0 + e1 * e1;
        for (int o = 16; o > 0; o >>= 1) sq += __shfl_xor(sq, o, 32);
        float nrm = ss * rsqrtf(sq + 1e-6f);
        e0 *= nrm; e1 *= nrm;
        float partner = __shfl_xor(e0, 8, 32);
        float out0 = e0;
        if (lane < 16) {
            int j = lane & 7;
            // freqs[h][j] = exp(log(pi) + (j*16+h) * (log(10pi)-log(pi))/128)
            float fr = __expf(1.14472988584940017f +
                              (float)(j * 16 + h) * 0.017988946038703484f);
            float th = p * fr;
            float c = __cosf(th), sn = __sinf(th);
            out0 = (lane < 8) ? (e0 * c - partner * sn)
                              : (e0 * c + partner * sn);
        }
        unsigned short* dst =
            (t == 0 ? q : k) + ((size_t)(b * NHEADS + h) * SEQ + s) * DHEAD;
        dst[lane]      = f2bf(out0);
        dst[lane + 32] = f2bf(e1);
    }
    {   // v passthrough, transposed
        const unsigned short* src = base + 2 * DMODEL;
        size_t vb = (size_t)(b * NHEADS + h) * DHEAD * SEQ;
        vT[vb + (size_t)lane * SEQ + s]        = src[lane];
        vT[vb + (size_t)(lane + 32) * SEQ + s] = src[lane + 32];
    }
}

// Flash attention. Block = 8 waves = 128 query rows of one (b,h).
// K/V tiles (32 keys) are staged into LDS once per block — via the Tensor
// Data Mover when available (wave 0 issues tensor_load_to_lds, waits on
// TENSORcnt, block barrier releases) — and all waves build WMMA fragments
// from LDS.
__global__ void k_attention(const unsigned short* __restrict__ q,
                            const unsigned short* __restrict__ k,
                            const unsigned short* __restrict__ vT,
                            unsigned short* __restrict__ attn) {
    __shared__ __align__(16) unsigned short kbuf[32 * DHEAD];  // [key][e]  4KB
    __shared__ __align__(16) unsigned short vbuf[DHEAD * 32];  // [e][key]  4KB
    __shared__ __align__(16) unsigned short pbuf[8][16 * 32];  //           8KB
    int tid = threadIdx.x;
    int warp = tid >> 5, lane = tid & 31;
    int b = blockIdx.z, h = blockIdx.y;
    int q0 = (blockIdx.x * 8 + warp) * 16;
    size_t hb = (size_t)(b * NHEADS + h) * SEQ * DHEAD;
    const unsigned short* qh = q  + hb;
    const unsigned short* kh = k  + hb;
    const unsigned short* vh = vT + hb;     // [DHEAD][SEQ]

    v16bf qa = load_a_frag(qh, DHEAD, q0, 0,  lane);
    v16bf qb = load_a_frag(qh, DHEAD, q0, 32, lane);
    v8f o0 = {}, o1 = {}, o2 = {}, o3 = {};
    float m[8], l[8];
    #pragma unroll
    for (int j = 0; j < 8; j++) { m[j] = -1e30f; l[j] = 0.f; }
    unsigned short* pw = pbuf[warp];

#ifdef HAVE_TDM
    unsigned koff = (unsigned)(unsigned long long)(kbuf);  // LDS byte offset
    unsigned voff = (unsigned)(unsigned long long)(vbuf);  // (addr[31:0])
#else
    // per-thread cooperative staging coordinates
    int kr = tid >> 3, kc = (tid & 7) * 8;  // K tile: 32 rows x 8 chunks
    int ve = tid >> 2, vc = (tid & 3) * 8;  // V tile: 64 rows x 4 chunks
#endif

    for (int j0 = 0; j0 < SEQ; j0 += 32) {
        // ---- stage K (32x64) and V (64x32) tiles into LDS ----
#ifdef HAVE_TDM
        if (warp == 0) {
            tdm_load_2d(kh + (size_t)j0 * DHEAD, koff, DHEAD, 32, DHEAD);
            tdm_load_2d(vh + j0,                 voff, 32, DHEAD, SEQ);
            __builtin_amdgcn_s_wait_tensorcnt(0);
        }
#else
        *(uint4*)(kbuf + kr * DHEAD + kc) =
            *(const uint4*)(kh + (size_t)(j0 + kr) * DHEAD + kc);
        *(uint4*)(vbuf + ve * 32 + vc) =
            *(const uint4*)(vh + (size_t)ve * SEQ + j0 + vc);
        if (j0 + 32 < SEQ) {  // prefetch next tiles into cache
            __builtin_prefetch(kh + (size_t)(j0 + 32 + kr) * DHEAD + kc, 0, 1);
            __builtin_prefetch(vh + (size_t)ve * SEQ + j0 + 32 + vc, 0, 1);
        }
#endif
        __syncthreads();

        // ---- scores: Q (16x64) x K^T (64x32) via LDS fragments ----
        v16bf kf0 = load_b_frag(kbuf, DHEAD, 0,  0,  lane);
        v16bf kf1 = load_b_frag(kbuf, DHEAD, 0,  32, lane);
        v16bf kf2 = load_b_frag(kbuf, DHEAD, 16, 0,  lane);
        v16bf kf3 = load_b_frag(kbuf, DHEAD, 16, 32, lane);
        v8f s0 = {}, s1 = {};
        s0 = wmma_bf16(qa, kf0, s0);
        s0 = wmma_bf16(qb, kf1, s0);
        s1 = wmma_bf16(qa, kf2, s1);
        s1 = wmma_bf16(qb, kf3, s1);

        // ---- online softmax, half-wave row reductions ----
        int half = lane >> 4, col = lane & 15;
        #pragma unroll
        for (int j = 0; j < 8; j++) {       // row = j + half*8
            float v0 = s0[j], v1 = s1[j];
            float mx = fmaxf(v0, v1);
            for (int o = 8; o > 0; o >>= 1) mx = fmaxf(mx, __shfl_xor(mx, o, 32));
            float mn = fmaxf(m[j], mx);
            float alpha = __expf(m[j] - mn);
            float p0 = __expf(v0 - mn), p1 = __expf(v1 - mn);
            float rs = p0 + p1;
            for (int o = 8; o > 0; o >>= 1) rs += __shfl_xor(rs, o, 32);
            l[j] = l[j] * alpha + rs;
            m[j] = mn;
            o0[j] *= alpha; o1[j] *= alpha; o2[j] *= alpha; o3[j] *= alpha;
            int row = j + half * 8;
            pw[row * 32 + col]      = f2bf(p0);
            pw[row * 32 + col + 16] = f2bf(p1);
        }
        asm volatile("s_wait_dscnt 0" ::: "memory");   // LDS relayout fence

        FragB16 pf;                       // P as A-fragment (16x32)
        {
            int r = lane & 15, kb = (lane >> 4) << 4;
            pf.u4[0] = *(const uint4*)(pw + r * 32 + kb);
            pf.u4[1] = *(const uint4*)(pw + r * 32 + kb + 8);
        }
        // ---- O += P (16x32) x V (32x64) via LDS fragments ----
        v16bf vf0 = load_b_frag(vbuf, 32, 0,  0, lane);
        v16bf vf1 = load_b_frag(vbuf, 32, 16, 0, lane);
        v16bf vf2 = load_b_frag(vbuf, 32, 32, 0, lane);
        v16bf vf3 = load_b_frag(vbuf, 32, 48, 0, lane);
        o0 = wmma_bf16(pf.v, vf0, o0);
        o1 = wmma_bf16(pf.v, vf1, o1);
        o2 = wmma_bf16(pf.v, vf2, o2);
        o3 = wmma_bf16(pf.v, vf3, o3);
        __syncthreads();   // protect kbuf/vbuf before next stage
    }

    int half = lane >> 4, col = lane & 15;
    #pragma unroll
    for (int j = 0; j < 8; j++) {
        float inv = 1.f / l[j];
        int row = q0 + j + half * 8;
        size_t rb = (size_t)(b * SEQ + row) * DMODEL + h * DHEAD;
        attn[rb + col]      = f2bf(o0[j] * inv);
        attn[rb + 16 + col] = f2bf(o1[j] * inv);
        attn[rb + 32 + col] = f2bf(o2[j] * inv);
        attn[rb + 48 + col] = f2bf(o3[j] * inv);
    }
}

// out = attn @ w_out^T + x (residual), f32 output. One wave per 16x64 strip.
__global__ void k_out_gemm(const unsigned short* __restrict__ A,
                           const unsigned short* __restrict__ W,
                           const float* __restrict__ x,
                           float* __restrict__ out) {
    int wave = (blockIdx.x * blockDim.x + threadIdx.x) >> 5;
    int lane = threadIdx.x & 31;
    const int n4 = DMODEL >> 6;            // 16
    int mt = wave / n4, ng = wave % n4;
    int n0 = ng * 64;
    v8f a0 = {}, a1 = {}, a2 = {}, a3 = {};
    #pragma unroll 4
    for (int k0 = 0; k0 < DMODEL; k0 += 32) {
        if (k0 + 64 < DMODEL) {
            __builtin_prefetch(A + (size_t)(mt * 16 + (lane & 15)) * DMODEL + k0 + 64, 0, 1);
        }
        v16bf a  = load_a_frag(A, DMODEL, mt * 16, k0, lane);
        v16bf b0 = load_b_frag(W, DMODEL, n0,      k0, lane);
        v16bf b1 = load_b_frag(W, DMODEL, n0 + 16, k0, lane);
        v16bf b2 = load_b_frag(W, DMODEL, n0 + 32, k0, lane);
        v16bf b3 = load_b_frag(W, DMODEL, n0 + 48, k0, lane);
        a0 = wmma_bf16(a, b0, a0);
        a1 = wmma_bf16(a, b1, a1);
        a2 = wmma_bf16(a, b2, a2);
        a3 = wmma_bf16(a, b3, a3);
    }
    int half = lane >> 4, col = n0 + (lane & 15);
    #pragma unroll
    for (int j = 0; j < 8; j++) {
        size_t row = (size_t)(mt * 16 + j + half * 8);
        out[row * DMODEL + col]      = a0[j] + x[row * DMODEL + col];
        out[row * DMODEL + col + 16] = a1[j] + x[row * DMODEL + col + 16];
        out[row * DMODEL + col + 32] = a2[j] + x[row * DMODEL + col + 32];
        out[row * DMODEL + col + 48] = a3[j] + x[row * DMODEL + col + 48];
    }
}

// ---------------------------------------------------------------- launcher ---

extern "C" void kernel_launch(void* const* d_in, const int* in_sizes, int n_in,
                              void* d_out, int out_size, void* d_ws, size_t ws_size,
                              hipStream_t stream) {
    (void)in_sizes; (void)n_in; (void)out_size; (void)ws_size;
    const float* x      = (const float*)d_in[0];
    const float* pos    = (const float*)d_in[1];
    const float* cond   = (const float*)d_in[2];
    const float* w_norm = (const float*)d_in[3];
    const float* w_qkv  = (const float*)d_in[4];
    const float* scale  = (const float*)d_in[5];
    const float* w_out  = (const float*)d_in[6];
    float* out = (float*)d_out;

    char* ws = (char*)d_ws;
    size_t off = 0;
    float* cs = (float*)(ws + off);              off += (size_t)BATCH * DMODEL * 4;
    off = (off + 255) & ~(size_t)255;
    unsigned short* wqkv_bf = (unsigned short*)(ws + off); off += (size_t)NQKV * DMODEL * 2;
    unsigned short* wout_bf = (unsigned short*)(ws + off); off += (size_t)DMODEL * DMODEL * 2;
    unsigned short* xn      = (unsigned short*)(ws + off); off += (size_t)ROWS * DMODEL * 2;
    unsigned short* qkv     = (unsigned short*)(ws + off); off += (size_t)ROWS * NQKV * 2;
    unsigned short* qb      = (unsigned short*)(ws + off); off += (size_t)ROWS * DMODEL * 2;
    unsigned short* kb      = (unsigned short*)(ws + off); off += (size_t)ROWS * DMODEL * 2;
    unsigned short* vTb     = (unsigned short*)(ws + off); off += (size_t)ROWS * DMODEL * 2;
    unsigned short* attn    = (unsigned short*)(ws + off); off += (size_t)ROWS * DMODEL * 2;

    k_cond_scale<<<(BATCH * DMODEL + 255) / 256, 256, 0, stream>>>(cond, w_norm, cs);
    k_f32_to_bf16<<<(NQKV * DMODEL + 255) / 256, 256, 0, stream>>>(w_qkv, wqkv_bf, NQKV * DMODEL);
    k_f32_to_bf16<<<(DMODEL * DMODEL + 255) / 256, 256, 0, stream>>>(w_out, wout_bf, DMODEL * DMODEL);
    k_rmsnorm<<<ROWS, 256, 0, stream>>>(x, cs, xn);

    // QKV GEMM: (ROWS/16)*(NQKV/64) = 256*48 = 12288 waves, 8 waves/block
    k_gemm_bf16<<<(ROWS / 16) * (NQKV / 64) / 8, 256, 0, stream>>>(
        xn, wqkv_bf, qkv, ROWS, NQKV, DMODEL);

    k_qkv_norm_rope<<<(ROWS * NHEADS) / 8, 256, 0, stream>>>(
        qkv, pos, scale, qb, kb, vTb);

    k_attention<<<dim3(SEQ / 16 / 8, NHEADS, BATCH), 256, 0, stream>>>(
        qb, kb, vTb, attn);

    // Out GEMM: (ROWS/16)*(DMODEL/64) = 256*16 = 4096 waves, 8 waves/block
    k_out_gemm<<<(ROWS / 16) * (DMODEL / 64) / 8, 256, 0, stream>>>(
        attn, wout_bf, x, out);
}
